// Module_RAIN_kuramotoV2_22823456211806
// MI455X (gfx1250) — matrix operand order repaired and checked by reading, not compiled, not run on hardware
//
#include <hip/hip_runtime.h>
#include <math.h>

// ---------------- CDNA5 WMMA types ----------------
typedef __attribute__((ext_vector_type(16))) _Float16 v16h;
typedef __attribute__((ext_vector_type(8)))  _Float16 v8h;
typedef __attribute__((ext_vector_type(8)))  float    v8f;

__device__ __forceinline__ v8f wmma_16x16x32_f16(v16h a, v16h b, v8f c) {
  // D(16x16,f32) = A(16x32,f16) * B(32x16,f16) + C
  return __builtin_amdgcn_wmma_f32_16x16x32_f16(false, a, false, b, (short)0, c, false, false);
}

__device__ __forceinline__ v16h cat8(v8h lo, v8h hi) {
  return __builtin_shufflevector(lo, hi, 0, 1, 2, 3, 4, 5, 6, 7,
                                 8, 9, 10, 11, 12, 13, 14, 15);
}
// A fragment: lane holds 16 contiguous f16 at p[0..15]
__device__ __forceinline__ v16h load_afrag(const _Float16* p) {
  const v8h* vp = (const v8h*)p;
  return cat8(vp[0], vp[1]);
}
// packed B fragment: 16 contiguous f16 per lane
__device__ __forceinline__ v16h load_bfrag(const _Float16* p) {
  const v8h* vp = (const v8h*)p;
  return cat8(vp[0], vp[1]);
}

__device__ __forceinline__ float sigf(float x) { return 1.0f / (1.0f + expf(-x)); }

// Problem constants
#define Bc   8
#define Lc   64
#define Ac   128
#define Dc   128
#define Hc   8
#define HDc  16
#define BA   1024      // B*A
#define T2   65536     // L*B*A tokens
#define NPAIR 131072   // B*A*A rows

#define GF_RELU   1
#define GF_OUTF16 2

// ---------------- weight pre-pack into WMMA B-fragment layout ----------------
// out[((kt*(N/16) + nt)*32 + lane)*16 + e] = (f16)W[(kt*32 + (lane<16?0:16) + e)*N + nt*16 + (lane&15)]
__global__ void k_packB(const float* __restrict__ W, _Float16* __restrict__ out,
                        int K, int N) {
  int idx = blockIdx.x * blockDim.x + threadIdx.x;
  if (idx >= K * N) return;
  int e    = idx & 15;
  int lane = (idx >> 4) & 31;
  int nt   = (idx >> 9) % (N >> 4);
  int kt   = idx / (512 * (N >> 4));
  int k = kt * 32 + ((lane < 16) ? 0 : 16) + e;
  int n = nt * 16 + (lane & 15);
  out[idx] = (_Float16)W[(size_t)k * N + n];
}

// ---------------- generic WMMA GEMM (packed-B weights) ----------------
// C(MxN) = A(MxK f16 row-major) @ Bp(packed) [+ A2 @ B2p] + bias; relu/f16-out flags.
// Block = 128 threads = 4 waves; each wave computes a 16(M) x 64(N) strip (4 accumulators).
__global__ __launch_bounds__(128)
void k_gemm(const _Float16* __restrict__ A, int lda,
            const _Float16* __restrict__ Bp,
            const float* __restrict__ bias,
            const _Float16* __restrict__ A2, int lda2,
            const _Float16* __restrict__ B2p, int K2,
            void* __restrict__ Cout, int ldc,
            int M, int N, int K, int flags) {
  const int lane = threadIdx.x & 31;
  const int wave = threadIdx.x >> 5;
  const int m0 = (blockIdx.x * 4 + wave) * 16;
  const int n0 = blockIdx.y * 64;
  if (m0 >= M) return;
  const int ntt = N >> 4;     // total n-tiles in packed B
  const int nt0 = n0 >> 4;

  v8f acc0 = {}, acc1 = {}, acc2 = {}, acc3 = {};
  // A fragment mapping (16x32 f16): lanes 0-15 -> rows, K {0..15}; lanes 16-31 -> K {16..31}
  const int arow = lane & 15;
  const int akb  = (lane < 16) ? 0 : 16;

  for (int k0 = 0; k0 < K; k0 += 32) {
    v16h af = load_afrag(A + (size_t)(m0 + arow) * lda + k0 + akb);
    const _Float16* fb = Bp + ((size_t)(k0 >> 5) * ntt + nt0) * 512 + lane * 16;
    acc0 = wmma_16x16x32_f16(af, load_bfrag(fb), acc0);
    acc1 = wmma_16x16x32_f16(af, load_bfrag(fb + 512), acc1);
    acc2 = wmma_16x16x32_f16(af, load_bfrag(fb + 1024), acc2);
    acc3 = wmma_16x16x32_f16(af, load_bfrag(fb + 1536), acc3);
  }
  if (A2) {
    for (int k0 = 0; k0 < K2; k0 += 32) {
      v16h af = load_afrag(A2 + (size_t)(m0 + arow) * lda2 + k0 + akb);
      const _Float16* fb = B2p + ((size_t)(k0 >> 5) * ntt + nt0) * 512 + lane * 16;
      acc0 = wmma_16x16x32_f16(af, load_bfrag(fb), acc0);
      acc1 = wmma_16x16x32_f16(af, load_bfrag(fb + 512), acc1);
      acc2 = wmma_16x16x32_f16(af, load_bfrag(fb + 1024), acc2);
      acc3 = wmma_16x16x32_f16(af, load_bfrag(fb + 1536), acc3);
    }
  }
  // C layout: lane col = lane&15 ; VGPR r -> row r + (lane<16 ? 0 : 8)
  const int crow = (lane < 16) ? 0 : 8;
  const int bcol = lane & 15;
  v8f accs[4] = {acc0, acc1, acc2, acc3};
#pragma unroll
  for (int t = 0; t < 4; ++t) {
    const int col = n0 + t * 16 + bcol;
    const float bv = bias ? bias[col] : 0.0f;
#pragma unroll
    for (int r = 0; r < 8; ++r) {
      float v = accs[t][r] + bv;
      if (flags & GF_RELU) v = fmaxf(v, 0.0f);
      const size_t o = (size_t)(m0 + crow + r) * ldc + col;
      if (flags & GF_OUTF16) ((_Float16*)Cout)[o] = (_Float16)v;
      else                   ((float*)Cout)[o] = v;
    }
  }
}

// ---------------- small kernels ----------------
// first MLP layer (2 -> 128, relu), writes f16.
// mode 0: token t reads x[t*2]; out token t
// mode 1: x_seq (B,L,A) -> out permuted to (L,B,A)
// mode 2: x0 = x_seq[:,0]; token t=(b,a) reads ((b*L)*A + a)*2
__global__ void k_lin2(const float* __restrict__ x, const float* __restrict__ W,
                       const float* __restrict__ b, _Float16* __restrict__ y,
                       int T, int mode) {
  int idx = blockIdx.x * blockDim.x + threadIdx.x;
  if (idx >= T * Dc) return;
  int t = idx >> 7, d = idx & 127;
  size_t xoff;
  int tout = t;
  if (mode == 1) {
    int a = t % Ac; int l = (t / Ac) % Lc; int bb = t / (Ac * Lc);
    xoff = (size_t)t * 2;
    tout = (l * Bc + bb) * Ac + a;
  } else if (mode == 2) {
    int a = t % Ac; int bb = t / Ac;
    xoff = ((size_t)bb * Lc * Ac + a) * 2;
  } else {
    xoff = (size_t)t * 2;
  }
  float v = x[xoff] * W[d] + x[xoff + 1] * W[Dc + d] + b[d];
  y[(size_t)tout * Dc + d] = (_Float16)fmaxf(v, 0.0f);
}

// LSTM gate nonlinearity: g (1024x512 f32) -> update c32, write h (f16), optional seq slice
__global__ void k_lstm_gate(const float* __restrict__ g, float* __restrict__ c32,
                            _Float16* __restrict__ h16, _Float16* __restrict__ seq) {
  int idx = blockIdx.x * blockDim.x + threadIdx.x;
  if (idx >= BA * Dc) return;
  int row = idx >> 7, d = idx & 127;
  const float* gr = g + (size_t)row * (4 * Dc);
  float gi = sigf(gr[d]);
  float gf = sigf(gr[Dc + d]);
  float gg = tanhf(gr[2 * Dc + d]);
  float go = sigf(gr[3 * Dc + d]);
  float c = gf * c32[idx] + gi * gg;
  float h = go * tanhf(c);
  c32[idx] = c;
  h16[idx] = (_Float16)h;
  if (seq) seq[idx] = (_Float16)h;
}

// Per-(b,h,i) fused temporal attention. q/k/v layout: ((l*B+b)*A + a)*D + h*16 + d.
// pair out: (b, i, j, h*16+d) f16
__global__ __launch_bounds__(256)
void k_attn(const _Float16* __restrict__ qb, const _Float16* __restrict__ kb,
            const _Float16* __restrict__ vb, _Float16* __restrict__ pair) {
  __shared__ _Float16 Ksh[Lc * HDc];      // [l][d]
  __shared__ _Float16 VshT[HDc * Lc];     // [d][l] (transposed -> contiguous B frags)
  __shared__ float    S[Ac * Lc];
  __shared__ _Float16 P[Ac * Lc];
  const int blk = blockIdx.x;
  const int i = blk % Ac;
  const int h = (blk / Ac) % Hc;
  const int b = blk / (Ac * Hc);
  const int tid = threadIdx.x;

  for (int e = tid; e < Lc * HDc; e += 256) {
    int l = e >> 4, d = e & 15;
    size_t src = ((size_t)(l * Bc + b) * Ac + i) * Dc + h * HDc + d;
    Ksh[e] = kb[src];
    VshT[d * Lc + l] = vb[src];
  }
  __syncthreads();

  const float scale = 0.08838834764831845f; // 1/sqrt(128)
  for (int e = tid; e < Ac * Lc; e += 256) {
    int j = e >> 6, l = e & 63;
    const _Float16* qp = qb + ((size_t)(l * Bc + b) * Ac + j) * Dc + h * HDc;
    float s = 0.0f;
#pragma unroll
    for (int d = 0; d < HDc; ++d) s += (float)qp[d] * (float)Ksh[l * HDc + d];
    S[e] = s * scale;
  }
  __syncthreads();

  if (tid < Ac) {                 // softmax over l per j-row
    float* row = S + tid * Lc;
    float m = row[0];
    for (int l = 1; l < Lc; ++l) m = fmaxf(m, row[l]);
    float sum = 0.0f;
    for (int l = 0; l < Lc; ++l) { float ev = expf(row[l] - m); row[l] = ev; sum += ev; }
    float inv = 1.0f / sum;
    for (int l = 0; l < Lc; ++l) P[tid * Lc + l] = (_Float16)(row[l] * inv);
  }
  __syncthreads();

  // Pair[j,d] = P(128x64) @ V(64x16) via WMMA; wave w owns j-tile w*16
  const int wave = tid >> 5, lane = tid & 31;
  v8f acc = {};
  const int arow = lane & 15, akb = (lane < 16) ? 0 : 16;
  const int bcol = lane & 15, bkb = (lane < 16) ? 0 : 16;
  for (int k0 = 0; k0 < Lc; k0 += 32) {
    v16h af = load_afrag(&P[(wave * 16 + arow) * Lc + k0 + akb]);
    v16h bf = load_bfrag(&VshT[bcol * Lc + k0 + bkb]);
    acc = wmma_16x16x32_f16(af, bf, acc);
  }
  const int crow = (lane < 16) ? 0 : 8;
  _Float16* dst = pair + ((size_t)b * Ac + i) * Ac * Dc;
#pragma unroll
  for (int r = 0; r < 8; ++r) {
    int j = wave * 16 + crow + r;
    dst[(size_t)j * Dc + h * HDc + bcol] = (_Float16)acc[r];
  }
}

// w_in[b,i,j,:] = [pair[b,i,j,:], pair[b,j,i,:]]
__global__ void k_pair_concat(const _Float16* __restrict__ pair, _Float16* __restrict__ cat) {
  size_t idx = (size_t)blockIdx.x * blockDim.x + threadIdx.x;
  if (idx >= (size_t)NPAIR * 256) return;
  int k = (int)(idx & 255);
  size_t row = idx >> 8;
  int j = (int)(row % Ac);
  int i = (int)((row / Ac) % Ac);
  int b = (int)(row / (Ac * Ac));
  _Float16 v;
  if (k < Dc) v = pair[(((size_t)b * Ac + i) * Ac + j) * Dc + k];
  else        v = pair[(((size_t)b * Ac + j) * Ac + i) * Dc + (k - Dc)];
  cat[idx] = v;
}

// intr input: [h[b,i,:], h[b,j,:]]
__global__ void k_h_concat(const _Float16* __restrict__ h16, _Float16* __restrict__ cat) {
  size_t idx = (size_t)blockIdx.x * blockDim.x + threadIdx.x;
  if (idx >= (size_t)NPAIR * 256) return;
  int k = (int)(idx & 255);
  size_t row = idx >> 8;
  int j = (int)(row % Ac);
  int i = (int)((row / Ac) % Ac);
  int b = (int)(row / (Ac * Ac));
  _Float16 v = (k < Dc) ? h16[((size_t)b * Ac + i) * Dc + k]
                        : h16[((size_t)b * Ac + j) * Dc + (k - Dc)];
  cat[idx] = v;
}

// att layer2 (128->1) + tanh + diag mask
__global__ void k_att_out(const _Float16* __restrict__ hid, const float* __restrict__ W2,
                          const float* __restrict__ b2, float* __restrict__ weight) {
  int row = blockIdx.x * blockDim.x + threadIdx.x;
  if (row >= NPAIR) return;
  int j = row % Ac, i = (row / Ac) % Ac;
  const _Float16* hp = hid + (size_t)row * Dc;
  float s = b2[0];
  for (int k = 0; k < Dc; ++k) s += (float)hp[k] * W2[k];
  weight[row] = (i == j) ? 0.0f : tanhf(s);
}

// agg[b,i,d] = sum_j weight[b,i,j]*intr[b,i,j,d] / (A-1)
__global__ void k_agg(const float* __restrict__ intr, const float* __restrict__ weight,
                      float* __restrict__ agg) {
  int bi = blockIdx.x;          // 0..1023
  int d = threadIdx.x;          // 0..63
  const float* wrow = weight + (size_t)bi * Ac;
  const float* ip = intr + (size_t)bi * Ac * 64 + d;
  float s = 0.0f;
  for (int j = 0; j < Ac; ++j) s += wrow[j] * ip[(size_t)j * 64];
  agg[bi * 64 + d] = s / 127.0f;
}

// mu/sig decoders (128->64 relu ->1); out[row,0]=mu, out[row,1]=sigmoid(sig)+1e-8
__global__ __launch_bounds__(64)
void k_decode(const float* __restrict__ sf, const float* __restrict__ agg,
              const float* __restrict__ muW1, const float* __restrict__ mub1,
              const float* __restrict__ muW2, const float* __restrict__ mub2,
              const float* __restrict__ sgW1, const float* __restrict__ sgb1,
              const float* __restrict__ sgW2, const float* __restrict__ sgb2,
              float* __restrict__ out) {
  int row = blockIdx.x, u = threadIdx.x;
  __shared__ float cc[128];
  __shared__ float hid[64];
  cc[u] = sf[row * 64 + u];
  cc[64 + u] = agg[row * 64 + u];
  __syncthreads();
  float s = mub1[u];
  for (int k = 0; k < 128; ++k) s += cc[k] * muW1[k * 64 + u];
  hid[u] = fmaxf(s, 0.0f);
  __syncthreads();
  if (u == 0) {
    float m = mub2[0];
    for (int t = 0; t < 64; ++t) m += hid[t] * muW2[t];
    out[row * 2 + 0] = m;
  }
  __syncthreads();
  s = sgb1[u];
  for (int k = 0; k < 128; ++k) s += cc[k] * sgW1[k * 64 + u];
  hid[u] = fmaxf(s, 0.0f);
  __syncthreads();
  if (u == 0) {
    float m = sgb2[0];
    for (int t = 0; t < 64; ++t) m += hid[t] * sgW2[t];
    out[row * 2 + 1] = sigf(m) + 1e-8f;
  }
}

// ---------------- orchestration ----------------
extern "C" void kernel_launch(void* const* d_in, const int* in_sizes, int n_in,
                              void* d_out, int out_size, void* d_ws, size_t ws_size,
                              hipStream_t stream) {
  (void)in_sizes; (void)n_in; (void)out_size; (void)ws_size;
  const float* xseq  = (const float*)d_in[0];
  const float* xlast = (const float*)d_in[1];
  auto P = [&](int i) { return (const float*)d_in[i]; };
  // params in setup_inputs insertion order; each MLP flattens as W1,b1,W2,b2
  const float *seW1=P(2),*seb1=P(3),*seW2=P(4),*seb2=P(5);
  const float *ihW1=P(6),*ihb1=P(7),*ihW2=P(8),*ihb2=P(9);
  const float *icW1=P(10),*icb1=P(11),*icW2=P(12),*icb2=P(13);
  const float *kW1=P(14),*kb1=P(15),*kW2=P(16),*kb2=P(17);
  const float *qW1=P(18),*qb1=P(19),*qW2=P(20),*qb2=P(21);
  const float *vW1=P(22),*vb1=P(23),*vW2=P(24),*vb2=P(25);
  const float *atW1=P(26),*atb1=P(27),*atW2=P(28),*atb2=P(29);
  const float *sfW1=P(30),*sfb1=P(31),*sfW2=P(32),*sfb2=P(33);
  const float *inW1=P(34),*inb1=P(35),*inW2=P(36),*inb2=P(37);
  const float *muW1=P(38),*mub1=P(39),*muW2=P(40),*mub2=P(41);
  const float *sgW1=P(42),*sgb1=P(43),*sgW2=P(44),*sgb2=P(45);
  const float *Wih=P(46),*Whh=P(47),*lb=P(48);

  // carve workspace
  char* base = (char*)d_ws;
  size_t off = 0;
  auto carve = [&](size_t bytes) -> char* {
    char* p = base + off;
    off += (bytes + 255) & ~(size_t)255;
    return p;
  };
  _Float16* xe   = (_Float16*)carve((size_t)T2 * Dc * 2);   // encoded seq, (L,B,A,D)
  _Float16* seq  = (_Float16*)carve((size_t)T2 * Dc * 2);   // lstm outputs, (L,B,A,D)
  _Float16* tmpb = (_Float16*)carve((size_t)T2 * Dc * 2);   // big scratch (layer1 hidden)
  _Float16* qbuf = (_Float16*)carve((size_t)T2 * Dc * 2);
  _Float16* kbuf = (_Float16*)carve((size_t)T2 * Dc * 2);
  _Float16* vbuf = (_Float16*)carve((size_t)T2 * Dc * 2);
  _Float16* pair = (_Float16*)carve((size_t)NPAIR * Dc * 2);
  _Float16* cat  = (_Float16*)carve((size_t)NPAIR * 256 * 2);
  _Float16* hid  = (_Float16*)carve((size_t)NPAIR * Dc * 2);
  float*    intr = (float*)   carve((size_t)NPAIR * 64 * 4);
  float*    g    = (float*)   carve((size_t)BA * 4 * Dc * 4);
  float*    c32  = (float*)   carve((size_t)BA * Dc * 4);
  _Float16* h16  = (_Float16*)carve((size_t)BA * Dc * 2);
  _Float16* t16  = (_Float16*)carve((size_t)BA * Dc * 2);   // small scratch
  _Float16* xd16 = (_Float16*)carve((size_t)BA * Dc * 2);
  float*    wgt  = (float*)   carve((size_t)NPAIR * 4);
  float*    sfb  = (float*)   carve((size_t)BA * 64 * 4);
  float*    aggb = (float*)   carve((size_t)BA * 64 * 4);
  // packed f16 weights (WMMA B-fragment layout)
  _Float16* seW2p=(_Float16*)carve(16384*2); _Float16* ihW2p=(_Float16*)carve(16384*2);
  _Float16* icW2p=(_Float16*)carve(16384*2);
  _Float16* kW1p=(_Float16*)carve(16384*2);  _Float16* kW2p=(_Float16*)carve(16384*2);
  _Float16* qW1p=(_Float16*)carve(16384*2);  _Float16* qW2p=(_Float16*)carve(16384*2);
  _Float16* vW1p=(_Float16*)carve(16384*2);  _Float16* vW2p=(_Float16*)carve(16384*2);
  _Float16* atW1p=(_Float16*)carve(32768*2);
  _Float16* sfW1p=(_Float16*)carve(16384*2); _Float16* sfW2p=(_Float16*)carve(8192*2);
  _Float16* inW1p=(_Float16*)carve(32768*2); _Float16* inW2p=(_Float16*)carve(8192*2);
  _Float16* WihP=(_Float16*)carve(65536*2);  _Float16* WhhP=(_Float16*)carve(65536*2);

  auto packB = [&](const float* s, _Float16* d, int K, int N) {
    k_packB<<<(K * N + 255) / 256, 256, 0, stream>>>(s, d, K, N);
  };
  packB(seW2, seW2p, Dc, Dc); packB(ihW2, ihW2p, Dc, Dc); packB(icW2, icW2p, Dc, Dc);
  packB(kW1, kW1p, Dc, Dc); packB(kW2, kW2p, Dc, Dc);
  packB(qW1, qW1p, Dc, Dc); packB(qW2, qW2p, Dc, Dc);
  packB(vW1, vW1p, Dc, Dc); packB(vW2, vW2p, Dc, Dc);
  packB(atW1, atW1p, 256, Dc);
  packB(sfW1, sfW1p, Dc, Dc); packB(sfW2, sfW2p, Dc, 64);
  packB(inW1, inW1p, 256, Dc); packB(inW2, inW2p, Dc, 64);
  packB(Wih, WihP, Dc, 4 * Dc); packB(Whh, WhhP, Dc, 4 * Dc);

  auto gemm = [&](const _Float16* Aa, int lda, const _Float16* Bp,
                  const float* bias, const _Float16* A2, int lda2,
                  const _Float16* B2p, int K2,
                  void* C, int ldc, int M, int N, int K, int flags) {
    dim3 gr(M / 64, N / 64);
    k_gemm<<<gr, 128, 0, stream>>>(Aa, lda, Bp, bias, A2, lda2, B2p, K2,
                                   C, ldc, M, N, K, flags);
  };

  // 1) state_enc on x_seq -> xe (permuted to (L,B,A,D))
  k_lin2<<<(T2 * Dc) / 256, 256, 0, stream>>>(xseq, seW1, seb1, tmpb, T2, 1);
  gemm(tmpb, Dc, seW2p, seb2, nullptr, 0, nullptr, 0, xe, Dc, T2, Dc, Dc, GF_OUTF16);

  // 2) init hidden / cell from x0
  k_lin2<<<(BA * Dc) / 256, 256, 0, stream>>>(xseq, ihW1, ihb1, t16, BA, 2);
  gemm(t16, Dc, ihW2p, ihb2, nullptr, 0, nullptr, 0, h16, Dc, BA, Dc, Dc, GF_OUTF16);
  k_lin2<<<(BA * Dc) / 256, 256, 0, stream>>>(xseq, icW1, icb1, t16, BA, 2);
  gemm(t16, Dc, icW2p, icb2, nullptr, 0, nullptr, 0, c32, Dc, BA, Dc, Dc, 0);

  // 3) LSTM scan over L (dual-product GEMM: x@Wih + h@Whh + b)
  for (int t = 0; t < Lc; ++t) {
    gemm(xe + (size_t)t * BA * Dc, Dc, WihP, lb,
         h16, Dc, WhhP, Dc, g, 4 * Dc, BA, 4 * Dc, Dc, 0);
    k_lstm_gate<<<(BA * Dc) / 256, 256, 0, stream>>>(g, c32, h16, seq + (size_t)t * BA * Dc);
  }

  // 4) k/q/v head MLPs over seq tokens
  gemm(seq, Dc, kW1p, kb1, nullptr, 0, nullptr, 0, tmpb, Dc, T2, Dc, Dc, GF_RELU | GF_OUTF16);
  gemm(tmpb, Dc, kW2p, kb2, nullptr, 0, nullptr, 0, kbuf, Dc, T2, Dc, Dc, GF_OUTF16);
  gemm(seq, Dc, qW1p, qb1, nullptr, 0, nullptr, 0, tmpb, Dc, T2, Dc, Dc, GF_RELU | GF_OUTF16);
  gemm(tmpb, Dc, qW2p, qb2, nullptr, 0, nullptr, 0, qbuf, Dc, T2, Dc, Dc, GF_OUTF16);
  gemm(seq, Dc, vW1p, vb1, nullptr, 0, nullptr, 0, tmpb, Dc, T2, Dc, Dc, GF_RELU | GF_OUTF16);
  gemm(tmpb, Dc, vW2p, vb2, nullptr, 0, nullptr, 0, vbuf, Dc, T2, Dc, Dc, GF_OUTF16);

  // 5) fused pairwise temporal attention
  k_attn<<<Bc * Hc * Ac, 256, 0, stream>>>(qbuf, kbuf, vbuf, pair);

  // 6) gat weight = tanh(att(concat(pair, pair^T))) * mask
  k_pair_concat<<<(unsigned)((size_t)NPAIR * 256 / 256), 256, 0, stream>>>(pair, cat);
  gemm(cat, 256, atW1p, atb1, nullptr, 0, nullptr, 0, hid, Dc, NPAIR, Dc, 256, GF_RELU | GF_OUTF16);
  k_att_out<<<NPAIR / 256, 256, 0, stream>>>(hid, atW2, atb2, wgt);

  // 7) decode step: one more LSTM step on state_enc(x_last)
  k_lin2<<<(BA * Dc) / 256, 256, 0, stream>>>(xlast, seW1, seb1, t16, BA, 0);
  gemm(t16, Dc, seW2p, seb2, nullptr, 0, nullptr, 0, xd16, Dc, BA, Dc, Dc, GF_OUTF16);
  gemm(xd16, Dc, WihP, lb, h16, Dc, WhhP, Dc, g, 4 * Dc, BA, 4 * Dc, Dc, 0);
  k_lstm_gate<<<(BA * Dc) / 256, 256, 0, stream>>>(g, c32, h16, nullptr);

  // 8) self_func
  gemm(h16, Dc, sfW1p, sfb1, nullptr, 0, nullptr, 0, t16, Dc, BA, Dc, Dc, GF_RELU | GF_OUTF16);
  gemm(t16, Dc, sfW2p, sfb2, nullptr, 0, nullptr, 0, sfb, 64, BA, 64, Dc, 0);

  // 9) intr_func + gat aggregation
  k_h_concat<<<(unsigned)((size_t)NPAIR * 256 / 256), 256, 0, stream>>>(h16, cat);
  gemm(cat, 256, inW1p, inb1, nullptr, 0, nullptr, 0, hid, Dc, NPAIR, Dc, 256, GF_RELU | GF_OUTF16);
  gemm(hid, Dc, inW2p, inb2, nullptr, 0, nullptr, 0, intr, 64, NPAIR, 64, Dc, 0);
  k_agg<<<BA, 64, 0, stream>>>(intr, wgt, aggb);

  // 10) mu/sigma decoders -> d_out (B,A,2) f32
  k_decode<<<BA, 64, 0, stream>>>(sfb, aggb, muW1, mub1, muW2, mub2,
                                  sgW1, sgb1, sgW2, sgb2, (float*)d_out);
}
